// ClusterAlignmentUnit_10634339025514
// MI455X (gfx1250) — compile-verified, matrix operand order checked by reading, not run on hardware
//
#include <hip/hip_runtime.h>
#include <hip/hip_bf16.h>
#include <math.h>

// ---------------------------------------------------------------------------
// CDNA5 (gfx1250) bf16 WMMA implementation of ClusterAlignmentUnit forward.
// wave32, v_wmma_f32_16x16x32_bf16 with f32 accumulation.
// Block tile 128x64, 8 waves (4x2), each wave owns a 32x32 tile (4 WMMAs/Kstep).
// ---------------------------------------------------------------------------

typedef __attribute__((ext_vector_type(16))) __bf16 v16bf;
typedef __attribute__((ext_vector_type(8)))  float  v8f;
typedef __attribute__((ext_vector_type(2)))  unsigned int u32x2;
typedef __attribute__((ext_vector_type(4)))  unsigned int u32x4;

#define NTHR 256

// ----------------------------- device helpers ------------------------------

__device__ __forceinline__ unsigned short f2bf(float f) {
  unsigned int u = __float_as_uint(f);
  unsigned int r = u + 0x7FFFu + ((u >> 16) & 1u);  // round-to-nearest-even
  return (unsigned short)(r >> 16);
}

__device__ __forceinline__ float geluf(float x) {
  return 0.5f * x * (1.0f + erff(x * 0.70710678118654752f));
}

__device__ __forceinline__ float blockReduceSum(float v, float* sh) {
  int t = threadIdx.x;
  sh[t] = v; __syncthreads();
  for (int s = blockDim.x >> 1; s > 0; s >>= 1) {
    if (t < s) sh[t] += sh[t + s];
    __syncthreads();
  }
  float r = sh[0]; __syncthreads();
  return r;
}

__device__ __forceinline__ float blockReduceMax(float v, float* sh) {
  int t = threadIdx.x;
  sh[t] = v; __syncthreads();
  for (int s = blockDim.x >> 1; s > 0; s >>= 1) {
    if (t < s) sh[t] = fmaxf(sh[t], sh[t + s]);
    __syncthreads();
  }
  float r = sh[0]; __syncthreads();
  return r;
}

__device__ __forceinline__ int pairInd(int a, int b) {
  if (a == b) return -1;
  int lo = a < b ? a : b;
  int hi = a < b ? b : a;
  return lo * 4 - lo * (lo - 1) / 2 + (hi - lo - 1);  // (0,1)->0 ... (3,4)->9
}

// ------------------------------ WMMA GEMM ----------------------------------
// C[z][M,N] = act( scale * A[z][M,K] @ op(W[z]) + bias + res[z] )
// A: bf16 row-major [M,K].  TRANSB=0 -> W is [K,N] row-major,
//                           TRANSB=1 -> W is [N,K] row-major (A @ W^T).
// Requirements exploited for wide loads: K % 4 == 0 and N % 4 == 0
// (true for every GEMM in this model: K,N in {64,512,900,1024,4096}).

#define BM 128
#define BN 64
#define BK 32
#define LDP (BK + 8)   // padded LDS row (u16 elems); 80B rows, 16B-aligned frags

union __align__(32) Frag {
  v16bf v;
  unsigned int u[8];
  u32x4 q[2];
};

template <int TRANSB>
__launch_bounds__(NTHR)
__global__ void k_gemm_bf16(const unsigned short* __restrict__ A, long long aStride,
                            const unsigned short* __restrict__ W, long long wStride,
                            const float* __restrict__ bias,
                            const float* __restrict__ res, long long resStride,
                            float* __restrict__ C, unsigned short* __restrict__ Cbf,
                            long long cStride,
                            int M, int N, int K, int act, float scale)
{
  int bz = blockIdx.z;
  A += (long long)bz * aStride;
  W += (long long)bz * wStride;
  const float* resb = res ? res + (long long)bz * resStride : nullptr;
  long long cOff = (long long)bz * cStride;

  __shared__ __align__(16) unsigned short sA[BM][LDP];
  __shared__ __align__(16) unsigned short sB[BN][LDP];   // [n][k]

  int tid  = threadIdx.x;
  int wave = tid >> 5;
  int lane = tid & 31;
  int m0 = blockIdx.y * BM;
  int n0 = blockIdx.x * BN;
  int wm = (wave & 3) * 32;   // wave tile M offset in block
  int wn = (wave >> 2) * 32;  // wave tile N offset in block

  v8f acc00 = {}, acc01 = {}, acc10 = {}, acc11 = {};

  int l15 = lane & 15;
  int akb = (lane >> 4) * 8;  // K sub-offset per lane half

  bool fullM = (m0 + BM) <= M;
  bool fullN = (n0 + BN) <= N;

  for (int k0 = 0; k0 < K; k0 += BK) {
    bool fullK = (k0 + BK) <= K;

    // ---------------- stage A tile: BM x BK (1024 u32x2, 4 per thread)
    if (fullM && fullK) {
      const unsigned short* ap = A + (long long)m0 * K + k0;
      #pragma unroll
      for (int j = 0; j < 4; ++j) {
        int q = tid + NTHR * j;
        int r = q >> 3;
        int c = (q & 7) * 4;
        const unsigned short* p = ap + (long long)r * K + c;
        *(u32x2*)&sA[r][c] = *(const u32x2*)p;
        if (j == 0 && (k0 + 2 * BK) <= K) __builtin_prefetch(p + BK, 0, 3);
      }
    } else {
      #pragma unroll
      for (int j = 0; j < 4; ++j) {
        int q = tid + NTHR * j;
        int r = q >> 3;
        int c = (q & 7) * 4;
        int gm = m0 + r;
        u32x2 v; v.x = 0; v.y = 0;
        if (gm < M) {
          const unsigned short* p = A + (long long)gm * K + k0 + c;
          if ((k0 + c + 4) <= K) {
            v = *(const u32x2*)p;
          } else {
            unsigned short e[4] = {0, 0, 0, 0};
            for (int i = 0; i < 4; ++i)
              if ((k0 + c + i) < K) e[i] = p[i];
            v.x = (unsigned int)e[0] | ((unsigned int)e[1] << 16);
            v.y = (unsigned int)e[2] | ((unsigned int)e[3] << 16);
          }
        }
        *(u32x2*)&sA[r][c] = v;
      }
    }

    // ---------------- stage B tile into sB[n][k]
    if (TRANSB) {
      // W is [N,K]; rows map directly (512 u32x2, 2 per thread)
      if (fullN && fullK) {
        const unsigned short* wp = W + (long long)n0 * K + k0;
        #pragma unroll
        for (int j = 0; j < 2; ++j) {
          int q = tid + NTHR * j;
          int r = q >> 3;
          int c = (q & 7) * 4;
          const unsigned short* p = wp + (long long)r * K + c;
          *(u32x2*)&sB[r][c] = *(const u32x2*)p;
          if (j == 0 && (k0 + 2 * BK) <= K) __builtin_prefetch(p + BK, 0, 3);
        }
      } else {
        #pragma unroll
        for (int j = 0; j < 2; ++j) {
          int q = tid + NTHR * j;
          int r = q >> 3;
          int c = (q & 7) * 4;
          int gn = n0 + r;
          u32x2 v; v.x = 0; v.y = 0;
          if (gn < N) {
            const unsigned short* p = W + (long long)gn * K + k0 + c;
            if ((k0 + c + 4) <= K) {
              v = *(const u32x2*)p;
            } else {
              unsigned short e[4] = {0, 0, 0, 0};
              for (int i = 0; i < 4; ++i)
                if ((k0 + c + i) < K) e[i] = p[i];
              v.x = (unsigned int)e[0] | ((unsigned int)e[1] << 16);
              v.y = (unsigned int)e[2] | ((unsigned int)e[3] << 16);
            }
          }
          *(u32x2*)&sB[r][c] = v;
        }
      }
    } else {
      // W is [K,N]; load contiguous along n, transpose into sB[n][k]
      // 512 u32x2 loads (4 n-values each), 2 per thread
      #pragma unroll
      for (int j = 0; j < 2; ++j) {
        int q = tid + NTHR * j;
        int kk = q >> 4;            // 0..31
        int n4 = (q & 15) * 4;      // 0..60
        u32x2 v; v.x = 0; v.y = 0;
        bool kin = (k0 + kk) < K;
        if (kin) {
          const unsigned short* p = W + (long long)(k0 + kk) * N + n0 + n4;
          if (fullN) {
            v = *(const u32x2*)p;
          } else {
            unsigned short e[4] = {0, 0, 0, 0};
            for (int i = 0; i < 4; ++i)
              if ((n0 + n4 + i) < N) e[i] = p[i];
            v.x = (unsigned int)e[0] | ((unsigned int)e[1] << 16);
            v.y = (unsigned int)e[2] | ((unsigned int)e[3] << 16);
          }
        }
        sB[n4 + 0][kk] = (unsigned short)(v.x & 0xFFFFu);
        sB[n4 + 1][kk] = (unsigned short)(v.x >> 16);
        sB[n4 + 2][kk] = (unsigned short)(v.y & 0xFFFFu);
        sB[n4 + 3][kk] = (unsigned short)(v.y >> 16);
      }
    }
    __syncthreads();

    // ---------------- fragments + 4 WMMAs (2x2 wave tile)
    Frag af0, af1, bf0, bf1;
    {
      int ar0 = wm + l15, ar1 = wm + 16 + l15;
      int br0 = wn + l15, br1 = wn + 16 + l15;
      af0.q[0] = *(const u32x4*)&sA[ar0][akb];
      af0.q[1] = *(const u32x4*)&sA[ar0][16 + akb];
      af1.q[0] = *(const u32x4*)&sA[ar1][akb];
      af1.q[1] = *(const u32x4*)&sA[ar1][16 + akb];
      bf0.q[0] = *(const u32x4*)&sB[br0][akb];
      bf0.q[1] = *(const u32x4*)&sB[br0][16 + akb];
      bf1.q[0] = *(const u32x4*)&sB[br1][akb];
      bf1.q[1] = *(const u32x4*)&sB[br1][16 + akb];
    }
    acc00 = __builtin_amdgcn_wmma_f32_16x16x32_bf16(false, af0.v, false, bf0.v,
                                                    (short)0, acc00, false, false);
    acc01 = __builtin_amdgcn_wmma_f32_16x16x32_bf16(false, af0.v, false, bf1.v,
                                                    (short)0, acc01, false, false);
    acc10 = __builtin_amdgcn_wmma_f32_16x16x32_bf16(false, af1.v, false, bf0.v,
                                                    (short)0, acc10, false, false);
    acc11 = __builtin_amdgcn_wmma_f32_16x16x32_bf16(false, af1.v, false, bf1.v,
                                                    (short)0, acc11, false, false);
    __syncthreads();
  }

  // ---------------- epilogue: 2x2 tiles; VGPR r -> M=r (+8 upper lanes), N=l15
  int lh8 = 8 * (lane >> 4);
  #pragma unroll
  for (int tm = 0; tm < 2; ++tm) {
    #pragma unroll
    for (int tn = 0; tn < 2; ++tn) {
      const v8f& a = tm == 0 ? (tn == 0 ? acc00 : acc01)
                             : (tn == 0 ? acc10 : acc11);
      int cn = n0 + wn + tn * 16 + l15;
      if (cn >= N) continue;
      #pragma unroll
      for (int r = 0; r < 8; ++r) {
        int cm = m0 + wm + tm * 16 + r + lh8;
        if (cm < M) {
          float x = a[r] * scale;
          if (bias) x += bias[cn];
          if (resb) x += resb[(long long)cm * N + cn];
          if (act == 1) x = geluf(x);
          long long o = cOff + (long long)cm * N + cn;
          if (C)   C[o]   = x;
          if (Cbf) Cbf[o] = f2bf(x);
        }
      }
    }
  }
}

// --------------------------- elementwise kernels ---------------------------

__launch_bounds__(NTHR)
__global__ void k_f2bf(const float* __restrict__ x, unsigned short* __restrict__ y, long long n) {
  long long i  = (long long)blockIdx.x * blockDim.x + threadIdx.x;
  long long st = (long long)gridDim.x * blockDim.x;
  for (; i < n; i += st) y[i] = f2bf(x[i]);
}

__launch_bounds__(NTHR)
__global__ void k_gelu_bf(const float* __restrict__ x, unsigned short* __restrict__ y, long long n) {
  long long i  = (long long)blockIdx.x * blockDim.x + threadIdx.x;
  long long st = (long long)gridDim.x * blockDim.x;
  for (; i < n; i += st) y[i] = f2bf(geluf(x[i]));
}

// y (and optional ybf) = LN(x); x may alias y.
__launch_bounds__(NTHR)
__global__ void k_layernorm(const float* x, const float* g, const float* bta, float eps,
                            int D, float* y, unsigned short* ybf) {
  __shared__ float sh[NTHR];
  long long row = blockIdx.x;
  const float* xr = x + row * (long long)D;
  int tid = threadIdx.x;
  float s = 0.f;
  for (int i = tid; i < D; i += blockDim.x) s += xr[i];
  float mean = blockReduceSum(s, sh) / (float)D;
  float vs = 0.f;
  for (int i = tid; i < D; i += blockDim.x) { float d = xr[i] - mean; vs += d * d; }
  float var = blockReduceSum(vs, sh) / (float)D;
  float inv = rsqrtf(var + eps);
  float* yr = y + row * (long long)D;
  unsigned short* ybr = ybf ? ybf + row * (long long)D : nullptr;
  for (int i = tid; i < D; i += blockDim.x) {
    float v = (xr[i] - mean) * inv * g[i] + bta[i];
    yr[i] = v;
    if (ybr) ybr[i] = f2bf(v);
  }
}

// e = LN(v + pos_emb[n+2] + type_emb), n = row % N; also bf16 mirror.
__launch_bounds__(NTHR)
__global__ void k_posln(const float* __restrict__ v, const float* __restrict__ pos,
                        const float* __restrict__ typ, const float* __restrict__ g,
                        const float* __restrict__ bta, float eps, int N, int D,
                        float* __restrict__ e, unsigned short* __restrict__ ebf) {
  __shared__ float sh[NTHR];
  long long row = blockIdx.x;
  int n = (int)(row % N);
  const float* vr = v + row * (long long)D;
  const float* pr = pos + (long long)(n + 2) * D;
  int tid = threadIdx.x;
  float s = 0.f;
  for (int i = tid; i < D; i += blockDim.x) s += vr[i] + pr[i] + typ[i];
  float mean = blockReduceSum(s, sh) / (float)D;
  float vs = 0.f;
  for (int i = tid; i < D; i += blockDim.x) {
    float d = vr[i] + pr[i] + typ[i] - mean; vs += d * d;
  }
  float var = blockReduceSum(vs, sh) / (float)D;
  float inv = rsqrtf(var + eps);
  float* er = e + row * (long long)D;
  unsigned short* ebr = ebf + row * (long long)D;
  for (int i = tid; i < D; i += blockDim.x) {
    float x = (vr[i] + pr[i] + typ[i] - mean) * inv * g[i] + bta[i];
    er[i] = x; ebr[i] = f2bf(x);
  }
}

// out = LN(vit + amt_b + v2), eps 1e-5
__launch_bounds__(NTHR)
__global__ void k_finalln(const float* __restrict__ vit, const float* __restrict__ amtb,
                          const float* __restrict__ v2, const float* __restrict__ g,
                          const float* __restrict__ bta, float eps, int D,
                          float* __restrict__ out) {
  __shared__ float sh[NTHR];
  long long row = blockIdx.x;
  const float* a = vit + row * (long long)D;
  const float* c = v2 + row * (long long)D;
  int tid = threadIdx.x;
  float s = 0.f;
  for (int i = tid; i < D; i += blockDim.x) s += a[i] + amtb[i] + c[i];
  float mean = blockReduceSum(s, sh) / (float)D;
  float vs = 0.f;
  for (int i = tid; i < D; i += blockDim.x) {
    float d = a[i] + amtb[i] + c[i] - mean; vs += d * d;
  }
  float var = blockReduceSum(vs, sh) / (float)D;
  float inv = rsqrtf(var + eps);
  float* o = out + row * (long long)D;
  for (int i = tid; i < D; i += blockDim.x)
    o[i] = (a[i] + amtb[i] + c[i] - mean) * inv * g[i] + bta[i];
}

// softmax over keys (scores [H,Sq,Sk] for one sequence unit), additive key mask,
// writes bf16 probabilities.  grid = (Sq, H)
__launch_bounds__(NTHR)
__global__ void k_softmax_bf16(const float* __restrict__ scores,
                               const float* __restrict__ maskRow, int Sk,
                               unsigned short* __restrict__ probs) {
  __shared__ float sh[NTHR];
  long long rowoff = ((long long)blockIdx.y * gridDim.x + blockIdx.x) * (long long)Sk;
  const float* r = scores + rowoff;
  int tid = threadIdx.x;
  float mx = -3.0e38f;
  for (int k = tid; k < Sk; k += blockDim.x) {
    float v = r[k] + (maskRow ? maskRow[k] : 0.f);
    mx = fmaxf(mx, v);
  }
  mx = blockReduceMax(mx, sh);
  float sum = 0.f;
  for (int k = tid; k < Sk; k += blockDim.x) {
    float v = r[k] + (maskRow ? maskRow[k] : 0.f);
    sum += expf(v - mx);
  }
  sum = blockReduceSum(sum, sh);
  float inv = 1.f / sum;
  unsigned short* p = probs + rowoff;
  for (int k = tid; k < Sk; k += blockDim.x) {
    float v = r[k] + (maskRow ? maskRow[k] : 0.f);
    p[k] = f2bf(expf(v - mx) * inv);
  }
}

// bf16 [U,S,D] -> bf16 [U*H,S,dh]
__launch_bounds__(NTHR)
__global__ void k_split_heads(const unsigned short* __restrict__ x,
                              unsigned short* __restrict__ y,
                              long long total, int S, int D, int H) {
  int dh = D / H;
  long long i  = (long long)blockIdx.x * blockDim.x + threadIdx.x;
  long long st = (long long)gridDim.x * blockDim.x;
  for (; i < total; i += st) {
    long long d = i % dh;
    long long t = i / dh;
    long long s = t % S; t /= S;
    long long h = t % H;
    long long u = t / H;
    y[i] = x[(u * S + s) * D + h * dh + d];
  }
}

// bf16 [U*H,S,dh] -> bf16 [U,S,D]
__launch_bounds__(NTHR)
__global__ void k_merge_heads(const unsigned short* __restrict__ x,
                              unsigned short* __restrict__ y,
                              long long total, int S, int D, int H) {
  int dh = D / H;
  long long i  = (long long)blockIdx.x * blockDim.x + threadIdx.x;
  long long st = (long long)gridDim.x * blockDim.x;
  for (; i < total; i += st) {
    long long dd = i % D;
    long long t = i / D;
    long long s = t % S;
    long long u = t / S;
    long long h = dd / dh;
    long long d = dd % dh;
    y[i] = x[((u * H + h) * S + s) * dh + d];
  }
}

// stable per-cluster pack -> seq [B*5,512,D] (f32 + bf16), additive mask (raw 0/1)
__launch_bounds__(NTHR)
__global__ void k_cluster_gather(const float* __restrict__ vit, const int* __restrict__ cl,
                                 int N, int D, float* __restrict__ seq,
                                 unsigned short* __restrict__ seqbf,
                                 float* __restrict__ maskAdd) {
  const int S = 512;
  int blk = blockIdx.x;
  int b = blk / 5, ci = blk % 5;
  __shared__ int idx[900];
  __shared__ int cntS;
  if (threadIdx.x == 0) {
    int c = 0;
    for (int n = 0; n < N; ++n)
      if (cl[b * N + n] == ci) idx[c++] = n;
    cntS = c;
  }
  __syncthreads();
  int count = cntS;
  int row0 = blockIdx.y * 64;
  for (int e = threadIdx.x; e < 64 * D; e += blockDim.x) {
    int s = row0 + e / D;
    int d = e % D;
    float v = 0.f;
    int src = s - 1;
    if (s >= 1 && src < count && src < 511)
      v = vit[((long long)b * N + idx[src]) * D + d];
    long long o = ((long long)blk * S + s) * D + d;
    seq[o] = v;
    seqbf[o] = f2bf(v);
  }
  if (blockIdx.y == 0)
    for (int k = threadIdx.x; k < S; k += blockDim.x)
      maskAdd[(long long)blk * S + k] = (k < count) ? 1.0f : 0.0f;
}

__launch_bounds__(NTHR)
__global__ void k_vmask(const int* __restrict__ cl, int j, long long n,
                        float* __restrict__ mask) {
  long long i  = (long long)blockIdx.x * blockDim.x + threadIdx.x;
  long long st = (long long)gridDim.x * blockDim.x;
  for (; i < n; i += st) mask[i] = (cl[i] == j) ? 0.f : -3.0e38f;
}

__launch_bounds__(NTHR)
__global__ void k_pool_copy(const float* __restrict__ vpout, int S, int D,
                            float* __restrict__ out2) {
  long long k = blockIdx.x;
  for (int d = threadIdx.x; d < D; d += blockDim.x)
    out2[k * D + d] = vpout[k * (long long)S * D + d];
}

// aw = softmax over batch dim of gelu(pooled) @ g2w_w + g2w_b    [B,3]
__launch_bounds__(128)
__global__ void k_aw(const float* __restrict__ pooled, const float* __restrict__ w,
                     const float* __restrict__ bia, int B, int D, float* __restrict__ aw) {
  __shared__ float logit[32 * 3];
  int t = threadIdx.x;
  if (t < B * 3) {
    int b = t / 3, c = t % 3;
    float s = bia[c];
    for (int d = 0; d < D; ++d) s += geluf(pooled[(long long)b * D + d]) * w[(long long)d * 3 + c];
    logit[t] = s;
  }
  __syncthreads();
  if (t < 3) {
    float mx = -3.0e38f;
    for (int b = 0; b < B; ++b) mx = fmaxf(mx, logit[b * 3 + t]);
    float sm = 0.f;
    for (int b = 0; b < B; ++b) sm += expf(logit[b * 3 + t] - mx);
    for (int b = 0; b < B; ++b) aw[b * 3 + t] = expf(logit[b * 3 + t] - mx) / sm;
  }
}

// 10-group masked softmax over sa rows + near/cross terms.  grid = (N, B)
__launch_bounds__(NTHR)
__global__ void k_align_state(const float* __restrict__ sa, const int* __restrict__ cl,
                              const float* __restrict__ aw, int N, float* __restrict__ out) {
  int n = blockIdx.x, b = blockIdx.y, tid = threadIdx.x;
  const float* row = sa + ((long long)b * N + n) * N;
  const int* crow = cl + (long long)b * N;
  int cn = crow[n];
  __shared__ float sh[NTHR];
  __shared__ float smax[10], ssum[10];
  float lmax[10], lsum[10];
  #pragma unroll
  for (int g = 0; g < 10; ++g) { lmax[g] = -3.0e38f; lsum[g] = 0.f; }
  for (int m = tid; m < N; m += blockDim.x) {
    if (m == n) continue;
    int ind = pairInd(cn, crow[m]);
    if (ind >= 0) lmax[ind] = fmaxf(lmax[ind], row[m]);
  }
  for (int g = 0; g < 10; ++g) {
    float r = blockReduceMax(lmax[g], sh);
    if (tid == 0) smax[g] = r;
  }
  __syncthreads();
  for (int m = tid; m < N; m += blockDim.x) {
    if (m == n) continue;
    int ind = pairInd(cn, crow[m]);
    if (ind >= 0) lsum[ind] += expf(row[m] - smax[ind]);
  }
  for (int g = 0; g < 10; ++g) {
    float r = blockReduceSum(lsum[g], sh);
    if (tid == 0) ssum[g] = r;
  }
  __syncthreads();
  float a0 = aw[b * 3 + 0], a1 = aw[b * 3 + 1], a2 = aw[b * 3 + 2];
  int r1 = n / 30, c1 = n % 30;
  for (int m = tid; m < N; m += blockDim.x) {
    float semv = 0.f;
    if (m != n) {
      int ind = pairInd(cn, crow[m]);
      if (ind >= 0) semv = expf(row[m] - smax[ind]) / ssum[ind];
    }
    int r2 = m / 30, c2 = m % 30;
    int dr = r1 - r2, dc = c1 - c2;
    float crossv = (m != n && (dr == 0 || dc == 0)) ? 1.f : 0.f;
    float nearv  = (m != n && (dr * dr + dc * dc) <= 9) ? 1.f : 0.f;
    out[((long long)b * N + n) * N + m] = semv * a0 + crossv * a1 + nearv * a2;
  }
}

// ------------------------------- host side ---------------------------------

static void* wsAlloc(char* base, size_t& off, size_t bytes) {
  void* r = base + off;
  off += (bytes + 255) & ~(size_t)255;
  return r;
}

static unsigned short* toBf(hipStream_t st, char* ws, size_t& off, const void* src, long long n) {
  unsigned short* d = (unsigned short*)wsAlloc(ws, off, (size_t)n * 2);
  long long blk = (n + NTHR * 4 - 1) / (NTHR * 4);
  if (blk < 1) blk = 1;
  if (blk > 16384) blk = 16384;
  k_f2bf<<<(int)blk, NTHR, 0, st>>>((const float*)src, d, n);
  return d;
}

static inline void gemm(hipStream_t st,
                        const unsigned short* A, long long aStr,
                        const unsigned short* W, long long wStr, int transB,
                        const float* bias, const float* res, long long resStr,
                        float* C, unsigned short* Cbf, long long cStr,
                        int M, int N, int K, int Z, int act, float scale) {
  dim3 g((N + BN - 1) / BN, (M + BM - 1) / BM, Z);
  if (transB)
    k_gemm_bf16<1><<<g, NTHR, 0, st>>>(A, aStr, W, wStr, bias, res, resStr,
                                       C, Cbf, cStr, M, N, K, act, scale);
  else
    k_gemm_bf16<0><<<g, NTHR, 0, st>>>(A, aStr, W, wStr, bias, res, resStr,
                                       C, Cbf, cStr, M, N, K, act, scale);
}

struct LayerW {
  const unsigned short *qw, *kw, *vw, *aow, *iw, *ow;
  const float *qb, *kb, *vb, *aob, *ib, *ob;
  const float *ln1g, *ln1b, *ln2g, *ln2b;
};

static LayerW makeLayer(hipStream_t st, char* ws, size_t& off, void* const* din,
                        int base, int D, int dff) {
  LayerW L;
  L.qw  = toBf(st, ws, off, din[base + 0], (long long)D * D);
  L.qb  = (const float*)din[base + 1];
  L.kw  = toBf(st, ws, off, din[base + 2], (long long)D * D);
  L.kb  = (const float*)din[base + 3];
  L.vw  = toBf(st, ws, off, din[base + 4], (long long)D * D);
  L.vb  = (const float*)din[base + 5];
  L.aow = toBf(st, ws, off, din[base + 6], (long long)D * D);
  L.aob = (const float*)din[base + 7];
  L.ln1g = (const float*)din[base + 8];
  L.ln1b = (const float*)din[base + 9];
  L.iw  = toBf(st, ws, off, din[base + 10], (long long)D * dff);
  L.ib  = (const float*)din[base + 11];
  L.ow  = toBf(st, ws, off, din[base + 12], (long long)dff * D);
  L.ob  = (const float*)din[base + 13];
  L.ln2g = (const float*)din[base + 14];
  L.ln2b = (const float*)din[base + 15];
  return L;
}

static void launch_copyish(hipStream_t st,
                           void (*kern)(const unsigned short*, unsigned short*, long long, int, int, int),
                           const unsigned short* x, unsigned short* y,
                           long long total, int S, int D, int H) {
  long long blk = (total + NTHR * 4 - 1) / (NTHR * 4);
  if (blk < 1) blk = 1;
  if (blk > 16384) blk = 16384;
  kern<<<(int)blk, NTHR, 0, st>>>(x, y, total, S, D, H);
}

// One full BertLayer: out = LN2( h + FFN(h) ), h = LN1( x + AO(MHA(x)) ).
static void bert_layer(hipStream_t st, char* ws, size_t& off,
                       const float* x, const unsigned short* xbf,
                       int U, int S, int D, int dff, int H,
                       const float* maskAdd, const LayerW& L, float* out) {
  size_t save = off;
  int dh = D / H;
  long long tok = (long long)U * S;
  long long nx  = tok * D;

  unsigned short* tmpbf  = (unsigned short*)wsAlloc(ws, off, (size_t)nx * 2);
  unsigned short* qh     = (unsigned short*)wsAlloc(ws, off, (size_t)nx * 2);
  unsigned short* kh     = (unsigned short*)wsAlloc(ws, off, (size_t)nx * 2);
  unsigned short* vh     = (unsigned short*)wsAlloc(ws, off, (size_t)nx * 2);
  unsigned short* ctxh   = (unsigned short*)wsAlloc(ws, off, (size_t)nx * 2);
  unsigned short* ctxm   = (unsigned short*)wsAlloc(ws, off, (size_t)nx * 2);
  float*  scores = (float*)wsAlloc(ws, off, (size_t)H * S * S * 4);
  unsigned short* probs  = (unsigned short*)wsAlloc(ws, off, (size_t)H * S * S * 2);
  float*  h      = (float*)wsAlloc(ws, off, (size_t)nx * 4);
  unsigned short* hbf    = (unsigned short*)wsAlloc(ws, off, (size_t)nx * 2);
  unsigned short* interbf = (unsigned short*)wsAlloc(ws, off, (size_t)tok * dff * 2);

  // Q, K, V projections (shared weights, M = U*S rows)
  gemm(st, xbf, 0, L.qw, 0, 0, L.qb, nullptr, 0, nullptr, tmpbf, 0, (int)tok, D, D, 1, 0, 1.f);
  launch_copyish(st, k_split_heads, tmpbf, qh, nx, S, D, H);
  gemm(st, xbf, 0, L.kw, 0, 0, L.kb, nullptr, 0, nullptr, tmpbf, 0, (int)tok, D, D, 1, 0, 1.f);
  launch_copyish(st, k_split_heads, tmpbf, kh, nx, S, D, H);
  gemm(st, xbf, 0, L.vw, 0, 0, L.vb, nullptr, 0, nullptr, tmpbf, 0, (int)tok, D, D, 1, 0, 1.f);
  launch_copyish(st, k_split_heads, tmpbf, vh, nx, S, D, H);

  float iscale = 1.0f / sqrtf((float)dh);
  for (int u = 0; u < U; ++u) {
    long long ho = (long long)u * H * S * dh;
    // scores = Q @ K^T * 1/sqrt(dh), batched over H heads
    gemm(st, qh + ho, (long long)S * dh, kh + ho, (long long)S * dh, 1,
         nullptr, nullptr, 0, scores, nullptr, (long long)S * S, S, S, dh, H, 0, iscale);
    k_softmax_bf16<<<dim3(S, H), NTHR, 0, st>>>(
        scores, maskAdd ? maskAdd + (long long)u * S : nullptr, S, probs);
    // ctx = probs @ V
    gemm(st, probs, (long long)S * S, vh + ho, (long long)S * dh, 0,
         nullptr, nullptr, 0, nullptr, ctxh + ho, (long long)S * dh, S, dh, S, H, 0, 1.f);
  }
  launch_copyish(st, k_merge_heads, ctxh, ctxm, nx, S, D, H);

  // attention output + residual, LN1
  gemm(st, ctxm, 0, L.aow, 0, 0, L.aob, x, 0, h, nullptr, 0, (int)tok, D, D, 1, 0, 1.f);
  k_layernorm<<<(int)tok, NTHR, 0, st>>>(h, L.ln1g, L.ln1b, 1e-12f, D, h, hbf);

  // FFN with fused GELU, residual, LN2
  gemm(st, hbf, 0, L.iw, 0, 0, L.ib, nullptr, 0, nullptr, interbf, 0, (int)tok, dff, D, 1, 1, 1.f);
  gemm(st, interbf, 0, L.ow, 0, 0, L.ob, h, 0, out, nullptr, 0, (int)tok, D, dff, 1, 0, 1.f);
  k_layernorm<<<(int)tok, NTHR, 0, st>>>(out, L.ln2g, L.ln2b, 1e-12f, D, out, nullptr);

  off = save;
}

// ------------------------------ entry point --------------------------------

enum {
  IN_VIT = 0, IN_CLUSTER = 1, IN_POOLED = 2, IN_TEXT = 3,
  P_AC1W = 4, P_AC1B, P_AC2W, P_AC2B, P_SEMW, P_SEMB, P_AMTW, P_AMTB,
  P_G2WW, P_G2WB, P_LNG, P_LNB,
  P_VP = 16,           // 16 leaves
  P_QWORD = 32, P_QPOS, P_QTYPE, P_QLNG, P_QLNB,
  P_QLAYER = 37,       // 16 leaves (dead for B>1)
  P_QPOOLW = 53, P_QPOOLB, P_PROJW, P_PROJB,
  P_VPOS = 57, P_VTYPE, P_VLNG, P_VLNB,
  P_VLAYER = 61        // 16 leaves
};

extern "C" void kernel_launch(void* const* d_in, const int* in_sizes, int n_in,
                              void* d_out, int out_size, void* d_ws, size_t ws_size,
                              hipStream_t stream) {
  (void)n_in; (void)out_size; (void)ws_size;
  const int N = 900, D = 1024, DFF = 4096, H = 16, SVP = 512;
  const float* vit = (const float*)d_in[IN_VIT];
  const int* cluster = (const int*)d_in[IN_CLUSTER];
  const float* pooled = (const float*)d_in[IN_POOLED];
  int B = in_sizes[IN_VIT] / (N * D);

  char* ws = (char*)d_ws;
  size_t off = 0;

  // persistent bf16 weights (reconverted every call; no caching)
  LayerW vp = makeLayer(stream, ws, off, d_in, P_VP, D, DFF);
  LayerW vl = makeLayer(stream, ws, off, d_in, P_VLAYER, D, DFF);
  unsigned short* semw = toBf(stream, ws, off, d_in[P_SEMW], (long long)D * D);

  float* outAligned = (float*)d_out;                       // [B,900,1024]
  float* outPool    = outAligned + (long long)B * N * D;   // [B,5,1024]
  float* outAlign   = outPool + (long long)B * 5 * D;      // [B,900,900]

  // ---- Phase 1: cluster pooler -> cluster_pool ----
  {
    size_t save = off;
    int U = B * 5;
    long long nseq = (long long)U * SVP * D;
    float* seq = (float*)wsAlloc(ws, off, (size_t)nseq * 4);
    unsigned short* seqbf = (unsigned short*)wsAlloc(ws, off, (size_t)nseq * 2);
    float* vpmask = (float*)wsAlloc(ws, off, (size_t)U * SVP * 4);
    float* vpout = (float*)wsAlloc(ws, off, (size_t)nseq * 4);
    k_cluster_gather<<<dim3(U, SVP / 64), NTHR, 0, stream>>>(vit, cluster, N, D,
                                                             seq, seqbf, vpmask);
    for (int c0 = 0; c0 < U; c0 += 5) {
      int uc = (U - c0 < 5) ? (U - c0) : 5;
      bert_layer(stream, ws, off,
                 seq + (long long)c0 * SVP * D, seqbf + (long long)c0 * SVP * D,
                 uc, SVP, D, DFF, H, vpmask + (long long)c0 * SVP, vp,
                 vpout + (long long)c0 * SVP * D);
    }
    k_pool_copy<<<U, NTHR, 0, stream>>>(vpout, SVP, D, outPool);
    off = save;
  }

  // ---- Phase 2: semantic alignment -> align_state ----
  {
    size_t save = off;
    long long nv = (long long)B * N * D;
    unsigned short* gbf = (unsigned short*)wsAlloc(ws, off, (size_t)nv * 2);
    unsigned short* sembf = (unsigned short*)wsAlloc(ws, off, (size_t)nv * 2);
    float* sa = (float*)wsAlloc(ws, off, (size_t)B * N * N * 4);
    float* aw = (float*)wsAlloc(ws, off, (size_t)B * 3 * 4);
    k_gelu_bf<<<4096, NTHR, 0, stream>>>(vit, gbf, nv);
    gemm(stream, gbf, 0, semw, 0, 0, (const float*)d_in[P_SEMB], nullptr, 0,
         nullptr, sembf, 0, B * N, D, D, 1, 0, 1.f);
    gemm(stream, sembf, (long long)N * D, sembf, (long long)N * D, 1,
         nullptr, nullptr, 0, sa, nullptr, (long long)N * N, N, N, D, B, 0, 1.f);
    k_aw<<<1, 128, 0, stream>>>(pooled, (const float*)d_in[P_G2WW],
                                (const float*)d_in[P_G2WB], B, D, aw);
    k_align_state<<<dim3(N, B), NTHR, 0, stream>>>(sa, cluster, aw, N, outAlign);
    off = save;
  }

  // ---- Phase 3: cluster reasoning (5x v_layer) + final LN -> aligned ----
  // B>1 => align_ind == 0 => alignment_token == amt_b (gelu(0)@W == 0).
  {
    size_t save = off;
    long long nv = (long long)B * N * D;
    float* e = (float*)wsAlloc(ws, off, (size_t)nv * 4);
    unsigned short* ebf = (unsigned short*)wsAlloc(ws, off, (size_t)nv * 2);
    float* va = (float*)wsAlloc(ws, off, (size_t)nv * 4);
    float* vb = (float*)wsAlloc(ws, off, (size_t)nv * 4);
    float* vmask = (float*)wsAlloc(ws, off, (size_t)B * N * 4);
    float* bufs[2] = { va, vb };
    const float* vcur = vit;
    for (int j = 0; j < 5; ++j) {
      k_vmask<<<64, NTHR, 0, stream>>>(cluster, j, (long long)B * N, vmask);
      k_posln<<<B * N, NTHR, 0, stream>>>(vcur, (const float*)d_in[P_VPOS],
                                          (const float*)d_in[P_VTYPE],
                                          (const float*)d_in[P_VLNG],
                                          (const float*)d_in[P_VLNB],
                                          1e-12f, N, D, e, ebf);
      float* vnext = bufs[j & 1];
      bert_layer(stream, ws, off, e, ebf, B, N, D, DFF, H, vmask, vl, vnext);
      vcur = vnext;
    }
    k_finalln<<<B * N, NTHR, 0, stream>>>(vit, (const float*)d_in[P_AMTB], vcur,
                                          (const float*)d_in[P_LNG],
                                          (const float*)d_in[P_LNB], 1e-5f, D,
                                          outAligned);
    off = save;
  }
}
